// TokenLevelModel_83966610637043
// MI455X (gfx1250) — matrix operand order repaired and checked by reading, not compiled, not run on hardware
//
#include <hip/hip_runtime.h>

// ---------------------------------------------------------------------------
// Problem constants (from the reference): B=96, S=128, C=141, D=768
// X = concat(h0,h1) @ W.T + b   -> GEMM  M=12288, K=1536, N=1536
// then char->token gather into (start_result, end_result), each (B,C,D) f32.
//
// GEMM is compute-bound (58 GFLOP vs ~15us of HBM time at 23.3 TB/s), so the
// kernel is organized around v_wmma_f32_16x16x32_bf16 issue rate:
//   - block tile 256x128, 8 waves, each wave 64x64 = 4x4 fragments
//     (16 WMMA per K-step, 1.0 ds_load_b128 per WMMA)
//   - double-buffered LDS, one barrier per K-step, global fetch of step k+1
//     issued before the 16 WMMAs of step k
//   - f32->bf16 conversion in-register (v_cvt_pk_bf16_f32 co-executes with
//     the XDL WMMA pipe on CDNA5)
// ---------------------------------------------------------------------------
namespace {
constexpr int kB = 96;
constexpr int kS = 128;
constexpr int kC = 141;
constexpr int kD = 768;
constexpr int kK = 2 * kD;   // 1536
constexpr int kN = 2 * kD;   // 1536
constexpr int kM = kB * kS;  // 12288
constexpr int TM = 256;      // block tile M
constexpr int TN = 128;      // block tile N
constexpr int KSTEP = 32;    // K per bf16 WMMA
}  // namespace

typedef __attribute__((ext_vector_type(16))) __bf16 v16bf;
typedef __attribute__((ext_vector_type(8)))  __bf16 v8bf;
typedef __attribute__((ext_vector_type(8)))  float  v8f;

__device__ __forceinline__ v8bf cvt8(const float4 a, const float4 b) {
  v8bf r;
  r[0] = (__bf16)a.x; r[1] = (__bf16)a.y; r[2] = (__bf16)a.z; r[3] = (__bf16)a.w;
  r[4] = (__bf16)b.x; r[5] = (__bf16)b.y; r[6] = (__bf16)b.z; r[7] = (__bf16)b.w;
  return r;
}

// ---------------------------------------------------------------------------
// GEMM:  X[m,n] = sum_k A[m,k] * W[n,k] + bias[n]
//   A[m,k] = (k < 768) ? h0[m, k] : h1[m, k-768]   (m = b*S + s)
// ---------------------------------------------------------------------------
__global__ __launch_bounds__(256, 1)
void gemm_bf16_wmma(const float* __restrict__ h0,
                    const float* __restrict__ h1,
                    const float* __restrict__ W,
                    const float* __restrict__ bias,
                    float* __restrict__ X)
{
  // Double-buffered tiles: 2*(16KB + 8KB) = 48 KB (WGP has 320 KB LDS).
  __shared__ __align__(64) __bf16 As[2][TM * KSTEP];  // [row m][k]
  __shared__ __align__(64) __bf16 Bs[2][TN * KSTEP];  // [row n][k]

  const int mTile = blockIdx.x * TM;
  const int nTile = blockIdx.y * TN;
  const int tid   = threadIdx.x;
  const int lane  = tid & 31;
  const int wid   = tid >> 5;
  const int wm    = (wid & 3) * 64;   // wave M offset inside tile (4 waves)
  const int wn    = (wid >> 2) * 64;  // wave N offset inside tile (2 waves)
  const int r     = lane & 15;        // fragment row/col within 16
  const int hh    = lane >> 4;        // half-wave select

  // A loader: one full 32-wide K chunk of one row per thread (256 rows).
  const float* __restrict__ a0 = h0 + (long)(mTile + tid) * kD;
  const float* __restrict__ a1 = h1 + (long)(mTile + tid) * kD;
  // B loader: half a 32-wide K chunk per thread (128 rows x 2).
  const int brow = tid >> 1;
  const int bcol = (tid & 1) * 16;
  const float* __restrict__ wr = W + (long)(nTile + brow) * kK;

  float4 aReg[8];
  float4 wReg[4];

  auto fetch = [&](int k0) {
    // 32-wide chunks never straddle the h0/h1 boundary (768 % 32 == 0).
    const float* ap = (k0 < kD) ? (a0 + k0) : (a1 + (k0 - kD));
#pragma unroll
    for (int i = 0; i < 8; ++i) aReg[i] = ((const float4*)ap)[i];
    const float* wp = wr + (k0 + bcol < kD ? 0 : 0) + k0 + bcol;  // W is contiguous in k
#pragma unroll
    for (int i = 0; i < 4; ++i) wReg[i] = ((const float4*)wp)[i];
  };

  auto stage = [&](int buf) {
    v8bf* ad = (v8bf*)&As[buf][tid * KSTEP];
#pragma unroll
    for (int i = 0; i < 4; ++i) ad[i] = cvt8(aReg[2 * i], aReg[2 * i + 1]);
    v8bf* wd = (v8bf*)&Bs[buf][brow * KSTEP + bcol];
    wd[0] = cvt8(wReg[0], wReg[1]);
    wd[1] = cvt8(wReg[2], wReg[3]);
  };

  v8f acc[4][4] = {};

  fetch(0);
  stage(0);
  __syncthreads();

  int cur = 0;
  for (int k0 = 0; k0 < kK; k0 += KSTEP) {
    const bool more = (k0 + KSTEP) < kK;
    if (more) fetch(k0 + KSTEP);  // global loads in flight during the WMMAs

    // B fragments (32x16 bf16): lane n (n<16) holds K=0..7,16..23 of column n,
    // lane n+16 holds K=8..15,24..31 (same order as A).
    v16bf bfrag[4];
#pragma unroll
    for (int g = 0; g < 4; ++g) {
      const __bf16* bp = &Bs[cur][(wn + g * 16 + r) * KSTEP];
      const v8bf blo = *(const v8bf*)(bp + hh * 8);
      const v8bf bhi = *(const v8bf*)(bp + 16 + hh * 8);
      bfrag[g] = __builtin_shufflevector(blo, bhi,
          0, 1, 2, 3, 4, 5, 6, 7, 8, 9, 10, 11, 12, 13, 14, 15);
    }

    // A fragments (16x32 bf16): lane r holds K=0..7,16..23; lane r+16 holds
    // K=8..15,24..31 of row (wm + f*16 + r).
#pragma unroll
    for (int f = 0; f < 4; ++f) {
      const __bf16* ap = &As[cur][(wm + f * 16 + r) * KSTEP];
      const v8bf lo = *(const v8bf*)(ap + hh * 8);
      const v8bf hi = *(const v8bf*)(ap + 16 + hh * 8);
      const v16bf afrag = __builtin_shufflevector(lo, hi,
          0, 1, 2, 3, 4, 5, 6, 7, 8, 9, 10, 11, 12, 13, 14, 15);
#pragma unroll
      for (int g = 0; g < 4; ++g)
        acc[f][g] = __builtin_amdgcn_wmma_f32_16x16x32_bf16(
            false, afrag, false, bfrag[g], (short)0, acc[f][g],
            false, false);
    }

    if (more) stage(cur ^ 1);  // write the other buffer; no conflict
    __syncthreads();           // single barrier per K-step
    cur ^= 1;
  }

  // Epilogue: f32 C/D layout — lanes 0-15: VGPR j -> M=j; lanes 16-31: M=j+8.
#pragma unroll
  for (int g = 0; g < 4; ++g) {
    const int col = nTile + wn + g * 16 + r;
    const float bv = bias[col];
#pragma unroll
    for (int f = 0; f < 4; ++f) {
      const int rowBase = mTile + wm + f * 16 + hh * 8;
#pragma unroll
      for (int j = 0; j < 8; ++j)
        X[(long)(rowBase + j) * kN + col] = acc[f][g][j] + bv;
    }
  }
}

// ---------------------------------------------------------------------------
// tokmap[b,c] = max{ s : starts[b,s] <= c < ends[b,s] }, else -1
// ---------------------------------------------------------------------------
__global__ __launch_bounds__(256)
void tokmap_kernel(const int* __restrict__ om, int* __restrict__ tokmap)
{
  __shared__ int so[2 * kS];
  const int b = blockIdx.x;
  if (threadIdx.x < 2 * kS) so[threadIdx.x] = om[b * 2 * kS + threadIdx.x];
  __syncthreads();
  for (int c = threadIdx.x; c < kC; c += blockDim.x) {
    int best = -1;
#pragma unroll 4
    for (int s = 0; s < kS; ++s) {
      if (c >= so[2 * s] && c < so[2 * s + 1]) best = s;  // increasing s => max
    }
    tokmap[b * kC + c] = best;
  }
}

// ---------------------------------------------------------------------------
// Gather: start_result[b,c,:] = X[b*S+t, 0:768], end = X[b*S+t, 768:1536]
// (zeros if t < 0).  One block per (b,c); 192 threads x float4 covers 768.
// X is L2-resident (75.5 MB < 192 MB L2), so this is cheap.
// ---------------------------------------------------------------------------
__global__ __launch_bounds__(192)
void gather_kernel(const float* __restrict__ X, const int* __restrict__ tokmap,
                   float* __restrict__ outS, float* __restrict__ outE)
{
  const int bc = blockIdx.x;
  const int b  = bc / kC;
  const int t  = tokmap[bc];
  const int i  = threadIdx.x;  // 0..191
  float4 vs = make_float4(0.f, 0.f, 0.f, 0.f);
  float4 ve = vs;
  if (t >= 0) {
    const float4* row = (const float4*)(X + (long)(b * kS + t) * kN);
    vs = row[i];
    ve = row[i + kD / 4];
  }
  ((float4*)(outS + (long)bc * kD))[i] = vs;
  ((float4*)(outE + (long)bc * kD))[i] = ve;
}

// ---------------------------------------------------------------------------
extern "C" void kernel_launch(void* const* d_in, const int* in_sizes, int n_in,
                              void* d_out, int out_size, void* d_ws, size_t ws_size,
                              hipStream_t stream)
{
  (void)in_sizes; (void)n_in; (void)out_size; (void)ws_size;

  const float* h0   = (const float*)d_in[0];  // (B,S,D) f32
  const float* h1   = (const float*)d_in[1];  // (B,S,D) f32
  const float* W    = (const float*)d_in[2];  // (2D,2D) f32
  const float* bias = (const float*)d_in[3];  // (2D,)   f32
  const int*   om   = (const int*)d_in[4];    // (B,S,2) i32

  // Workspace layout: X (kM*kN f32 = 75.5 MB) | tokmap (B*C i32)
  float* X      = (float*)d_ws;
  int*   tokmap = (int*)((char*)d_ws + (size_t)kM * kN * sizeof(float));

  dim3 g(kM / TM, kN / TN);  // 48 x 12
  gemm_bf16_wmma<<<g, 256, 0, stream>>>(h0, h1, W, bias, X);
  tokmap_kernel<<<kB, 256, 0, stream>>>(om, tokmap);
  gather_kernel<<<kB * kC, 192, 0, stream>>>(
      X, tokmap, (float*)d_out, (float*)d_out + (size_t)kB * kC * kD);
}
